// SkeletonTokenizer_34059090657339
// MI455X (gfx1250) — matrix operand order repaired and checked by reading, not compiled, not run on hardware
//
#include <hip/hip_runtime.h>
#include <hip/hip_bf16.h>

typedef __attribute__((ext_vector_type(16))) __bf16  v16bf;
typedef __attribute__((ext_vector_type(8)))  float   v8f;

#define KTOK 16384
#define DMODEL 1024

// ---------------------------------------------------------------------------
// Async global -> LDS copy (CDNA5, ASYNCcnt-tracked), 16 bytes per lane.
// the vdst VGPR holds the LDS byte offset; low 32 bits of a flat LDS pointer are the
// wave-relative LDS offset (ISA 10.2 aperture mapping).
// ---------------------------------------------------------------------------
__device__ __forceinline__ void async_copy_b128(unsigned ldsOff, const void* gp) {
    unsigned long long ga = (unsigned long long)(uintptr_t)gp;
    asm volatile("global_load_async_to_lds_b128 %0, %1, off"
                 :: "v"(ldsOff), "v"(ga) : "memory");
}
__device__ __forceinline__ void wait_async0() {
    asm volatile("s_wait_asynccnt 0x0" ::: "memory");
}

// ---------------------------------------------------------------------------
// Kernel 1: H = relu(X @ W1 + b1) -> bf16.  X:[K,3], W1:[3,512], H:[K,512]
// ---------------------------------------------------------------------------
__global__ void k_mlp1(const float* __restrict__ X, const float* __restrict__ W1,
                       const float* __restrict__ b1, __bf16* __restrict__ H) {
    int tid = blockIdx.x * blockDim.x + threadIdx.x;   // K*512 threads
    int k = tid >> 9, j = tid & 511;
    float x0 = X[k * 3 + 0], x1 = X[k * 3 + 1], x2 = X[k * 3 + 2];
    float acc = fmaf(x0, W1[j],
                fmaf(x1, W1[512 + j],
                fmaf(x2, W1[1024 + j], b1[j])));
    H[tid] = (__bf16)fmaxf(acc, 0.0f);
}

// ---------------------------------------------------------------------------
// Kernel 2: pack f32 weight W[Kd,N] (row-major) into bf16 WMMA-B-fragment
// order: tile (kt: 32 k's, nt: 16 n's) -> [lane 0..31][16 bf16], where
// lane = (khalf<<4)|ncol, element j holds K = kt*32 + khalf*16 + j.
// ---------------------------------------------------------------------------
__global__ void k_pack_b(const float* __restrict__ W, __bf16* __restrict__ Bp,
                         int Kd, int N) {
    int o = blockIdx.x * blockDim.x + threadIdx.x;     // Kd*N threads
    int j = o & 15;
    int lane = (o >> 4) & 31;
    int rest = o >> 9;
    int ntiles = N >> 4;
    int nt = rest % ntiles, kt = rest / ntiles;
    int n = (nt << 4) + (lane & 15);
    int k = (kt << 5) + ((lane >> 4) << 4) + j;
    Bp[o] = (__bf16)W[(size_t)k * N + n];
}

// ---------------------------------------------------------------------------
// Kernel 3: combined [K,4096] bf16 = [ J[k] | gamma(k) | J[p[k]] | gamma(p[k]) ]
// ---------------------------------------------------------------------------
__global__ void k_combined(const __bf16* __restrict__ J, const int* __restrict__ parent,
                           __bf16* __restrict__ Cmb) {
    int tid = blockIdx.x * blockDim.x + threadIdx.x;   // K*4096 threads
    int k = tid >> 12, c = tid & 4095;
    int seg = c >> 10, cc = c & 1023;
    __bf16 val;
    if (seg == 0) {
        val = J[(size_t)k * DMODEL + cc];
    } else if (seg == 2) {
        int p = parent[k];
        val = J[(size_t)p * DMODEL + cc];
    } else {
        int pos = (seg == 1) ? k : parent[k];
        int i = cc >> 1;
        float f = exp2f(-0.02595256324f * (float)i);   // 10000^(-2i/1024)
        float ang = (float)pos * f;
        float v = (cc & 1) ? __cosf(ang) : __sinf(ang);
        val = (__bf16)v;
    }
    Cmb[tid] = val;
}

// ---------------------------------------------------------------------------
// bf16 WMMA GEMM: C[M,N] = act(A[M,Kd] @ B[Kd,N] + bias)
// Block: 256 threads (8 waves), 64x256 output tile.
// Each wave: 2x4 grid of 16x16 WMMA tiles (32 rows x 64 cols).
// A tile (64x64 bf16 = 8KB) double-buffered in LDS via async DMA;
// 16 WMMAs per wave between barriers. B fragments streamed from L2.
// ---------------------------------------------------------------------------
template <typename OutT, bool RELU>
__global__ void k_gemm_bf16(const __bf16* __restrict__ A, const __bf16* __restrict__ Bp,
                            const float* __restrict__ bias, OutT* __restrict__ C,
                            int Kd, int N) {
    __shared__ __bf16 As[2][64 * 64];                  // 2 x 8 KB

    const int t     = threadIdx.x;
    const int lane  = t & 31;
    const int wave  = t >> 5;
    const int waveM = wave & 1;                        // 0..1 -> 32-row half
    const int waveN = wave >> 1;                       // 0..3 -> 64-col group
    const int rowBase = blockIdx.y << 6;               // 64 rows / block
    const int ntiles  = N >> 4;
    const int nt0     = (blockIdx.x << 4) + (waveN << 2);

    // A staging: 256 threads x 32 B = 8 KB (2 x b128 async per thread)
    const int r  = t >> 2;                             // 0..63
    const int kg = (t & 3) << 4;                       // 0,16,32,48 bf16
    const int rl    = (waveM << 5) + (lane & 15);      // frag rows rl, rl+16
    const int khalf = lane >> 4;

    v8f acc[2][4];
#pragma unroll
    for (int mi = 0; mi < 2; ++mi)
#pragma unroll
        for (int j = 0; j < 4; ++j) acc[mi][j] = {};

    const size_t aBase = (size_t)(rowBase + r) * Kd + kg;
    const int    ktEnd = Kd >> 6;                      // 64 k's per step

    const unsigned lo0 = (unsigned)(uintptr_t)(&As[0][r * 64 + kg]);
    const unsigned lo1 = (unsigned)(uintptr_t)(&As[1][r * 64 + kg]);

    // prologue: tile 0 -> buffer 0
    async_copy_b128(lo0,      A + aBase);
    async_copy_b128(lo0 + 16, A + aBase + 8);
    wait_async0();
    __syncthreads();

    const v16bf* BpV = reinterpret_cast<const v16bf*>(Bp);

    for (int kt = 0; kt < ktEnd; ++kt) {
        const int cur = kt & 1;
        if (kt + 1 < ktEnd) {
            const unsigned lo = cur ? lo0 : lo1;
            const __bf16* gp = A + aBase + ((size_t)(kt + 1) << 6);
            async_copy_b128(lo,      gp);
            async_copy_b128(lo + 16, gp + 8);
        }
        const uint* AsU = reinterpret_cast<const uint*>(As[cur]);

#pragma unroll
        for (int ks = 0; ks < 2; ++ks) {               // two 32-deep substeps
            // A fragments, contiguous k order: element s holds k = 16*khalf + s (2x ds_load_b128 each)
            union { v16bf v; uint u[8]; } a0, a1;
#pragma unroll
            for (int vv = 0; vv < 8; ++vv) {
                const int kpair = (ks << 4) + (khalf << 3) + vv;
                a0.u[vv] = AsU[(rl << 5) + kpair];
                a1.u[vv] = AsU[((rl + 16) << 5) + kpair];
            }
            const size_t bt = ((size_t)((kt << 1) + ks) * ntiles + nt0) << 5;
#pragma unroll
            for (int j = 0; j < 4; ++j) {
                const v16bf b = BpV[bt + (j << 5) + lane];
                acc[0][j] = __builtin_amdgcn_wmma_f32_16x16x32_bf16(
                    false, a0.v, false, b, (short)0, acc[0][j], false, false);
                acc[1][j] = __builtin_amdgcn_wmma_f32_16x16x32_bf16(
                    false, a1.v, false, b, (short)0, acc[1][j], false, false);
            }
        }
        wait_async0();        // next tile landed in LDS
        __syncthreads();      // all waves done reading As[cur]
    }

    // Epilogue: C/D layout -> VGPR i: row = i + 8*khalf, col = lane&15
    const int colBase = nt0 << 4;
#pragma unroll
    for (int mi = 0; mi < 2; ++mi) {
#pragma unroll
        for (int j = 0; j < 4; ++j) {
            const int col = colBase + (j << 4) + (lane & 15);
            const float bc = bias[col];
#pragma unroll
            for (int i = 0; i < 8; ++i) {
                const int row = rowBase + (waveM << 5) + (mi << 4) + i + (khalf << 3);
                float v = acc[mi][j][i] + bc;
                if (RELU) v = fmaxf(v, 0.0f);
                C[(size_t)row * N + col] = (OutT)v;
            }
        }
    }
}

// ---------------------------------------------------------------------------
extern "C" void kernel_launch(void* const* d_in, const int* in_sizes, int n_in,
                              void* d_out, int out_size, void* d_ws, size_t ws_size,
                              hipStream_t stream) {
    (void)in_sizes; (void)n_in; (void)out_size; (void)ws_size;
    const float* positions = (const float*)d_in[0];
    const int*   parent    = (const int*)  d_in[1];
    const float* W1 = (const float*)d_in[2];
    const float* b1 = (const float*)d_in[3];
    const float* W2 = (const float*)d_in[4];
    const float* b2 = (const float*)d_in[5];
    const float* W3 = (const float*)d_in[6];
    const float* b3 = (const float*)d_in[7];
    const float* W4 = (const float*)d_in[8];
    const float* b4 = (const float*)d_in[9];
    float* out = (float*)d_out;

    char* ws = (char*)d_ws;
    size_t off = 0;
    __bf16* Hbf = (__bf16*)(ws + off); off += (size_t)KTOK * 512  * 2;  // 16 MB
    __bf16* Jbf = (__bf16*)(ws + off); off += (size_t)KTOK * 1024 * 2;  // 32 MB
    __bf16* Cmb = (__bf16*)(ws + off); off += (size_t)KTOK * 4096 * 2;  // 128 MB
    __bf16* hbf = (__bf16*)(ws + off); off += (size_t)KTOK * 2048 * 2;  // 64 MB
    __bf16* W2p = (__bf16*)(ws + off); off += (size_t)512  * 1024 * 2;  // 1 MB
    __bf16* W3p = (__bf16*)(ws + off); off += (size_t)4096 * 2048 * 2;  // 16 MB
    __bf16* W4p = (__bf16*)(ws + off); off += (size_t)2048 * 1024 * 2;  // 4 MB

    // weight conversion / fragment packing
    k_pack_b<<<(512 * 1024) / 256, 256, 0, stream>>>(W2, W2p, 512, 1024);
    k_pack_b<<<(4096 * 2048) / 256, 256, 0, stream>>>(W3, W3p, 4096, 2048);
    k_pack_b<<<(2048 * 1024) / 256, 256, 0, stream>>>(W4, W4p, 2048, 1024);

    // layer 1 (tiny)
    k_mlp1<<<(KTOK * 512) / 256, 256, 0, stream>>>(positions, W1, b1, Hbf);

    // J = H @ W2 + b2  (bf16 out, no relu)
    k_gemm_bf16<__bf16, false><<<dim3(1024 / 256, KTOK / 64), 256, 0, stream>>>(
        Hbf, W2p, b2, Jbf, 512, 1024);

    // combined = [J | gamma(k) | J[parent] | gamma(parent)]
    k_combined<<<(KTOK * 4096) / 256, 256, 0, stream>>>(Jbf, parent, Cmb);

    // h = relu(combined @ W3 + b3)  (bf16 out)
    k_gemm_bf16<__bf16, true><<<dim3(2048 / 256, KTOK / 64), 256, 0, stream>>>(
        Cmb, W3p, b3, hbf, 4096, 2048);

    // out = h @ W4 + b4  (f32 out)
    k_gemm_bf16<float, false><<<dim3(1024 / 256, KTOK / 64), 256, 0, stream>>>(
        hbf, W4p, b4, out, 2048, 1024);
}
